// _GroupedGRULayer_30099130810779
// MI455X (gfx1250) — matrix operand order repaired and checked by reading, not compiled, not run on hardware
//
#include <hip/hip_runtime.h>

typedef __attribute__((ext_vector_type(16))) _Float16 v16h;
typedef __attribute__((ext_vector_type(8)))  float    v8f;

#define Bq   32
#define Tq   512
#define Gq   8
#define IGq  128
#define HGq  128
#define NOq  384      // 3*HG
#define HIDq 1024
#define INq  1024

// 32x16 (KxN) f16 B-fragment for V_WMMA_F32_16X16X32_F16 from fp32 weights
// stored row-major [O][K] (D = A * W^T -> B[k][n] = W[o_base+n][k]).
// Lane l: n = l%16, K = 8*(l/16)+j (j<8), 16+8*(l/16)+(j-8) (j>=8).
static __device__ __forceinline__ v16h load_bfrag(const float* __restrict__ wrow) {
  v16h b;
  const float* p = wrow - 8 * ((threadIdx.x & 31) >> 4);
#pragma unroll
  for (int j = 0; j < 8; ++j) { b[j] = (_Float16)p[j]; b[j + 8] = (_Float16)p[16 + j]; }
  return b;
}

// ---------------------------------------------------------------------------
// Phase 1: gx[g][t][b][o] (f16) = x_g[b,t,:] @ W_ih[g]^T + b_ih[g]
// Grid (64, 8): blockIdx.y = group; each block loops over 8 chunks of 32 rows.
// 8 waves: wave = (mtile<<2)|q, mtile picks 16 rows, q picks 6 ntiles.
// W_ih B-fragments register-resident across the 8 row-chunks.
// ---------------------------------------------------------------------------
__global__ __launch_bounds__(256)
void gru_gx_gemm(const float* __restrict__ x,
                 const float* __restrict__ W_ih,
                 const float* __restrict__ b_ih,
                 _Float16* __restrict__ gx) {
  const int g    = blockIdx.y;
  const int lane = threadIdx.x & 31;
  const int wave = threadIdx.x >> 5;
  const int hi   = lane >> 4;
  const int ln   = lane & 15;
  const int mtile = wave >> 2;
  const int q     = wave & 3;

  v16h Bf[6][4];
  float bia[6];
#pragma unroll
  for (int jn = 0; jn < 6; ++jn) {
    const int o = (q * 6 + jn) * 16 + ln;
    bia[jn] = b_ih[g * NOq + o];
#pragma unroll
    for (int ks = 0; ks < 4; ++ks)
      Bf[jn][ks] = load_bfrag(W_ih + ((size_t)g * NOq + o) * IGq + ks * 32 + hi * 16);
  }

  for (int i = 0; i < 8; ++i) {
    const int chunk = blockIdx.x * 8 + i;   // 0..511 -> 32 rows of M = b*T+t
    const int bb = chunk >> 4;
    const int t0 = (chunk & 15) * 32;

    v8f acc[6];
#pragma unroll
    for (int jn = 0; jn < 6; ++jn)
#pragma unroll
      for (int r = 0; r < 8; ++r) acc[jn][r] = bia[jn];

#pragma unroll
    for (int ks = 0; ks < 4; ++ks) {
      const int t = t0 + mtile * 16 + ln;
      const float* xr = x + (size_t)(bb * Tq + t) * INq + g * IGq + ks * 32 + hi * 8;
      v16h a;
#pragma unroll
      for (int j = 0; j < 8; ++j) {
        a[j]     = (_Float16)xr[j];
        a[j + 8] = (_Float16)xr[16 + j];
      }
#pragma unroll
      for (int jn = 0; jn < 6; ++jn)
        acc[jn] = __builtin_amdgcn_wmma_f32_16x16x32_f16(
            false, a, false, Bf[jn][ks], (short)0, acc[jn], false, false);
    }

#pragma unroll
    for (int jn = 0; jn < 6; ++jn) {
      const int o = (q * 6 + jn) * 16 + ln;
#pragma unroll
      for (int r = 0; r < 8; ++r) {
        const int t = t0 + mtile * 16 + hi * 8 + r;
        gx[(((size_t)g * Tq + t) * Bq + bb) * NOq + o] = (_Float16)acc[jn][r];
      }
    }
  }
}

// Issue six GLOBAL_LOAD_ASYNC_TO_LDS_B128 per thread: copies 96 B/thread
// (24576 B slab with 256 threads) from global gx into an LDS buffer,
// tracked by ASYNCcnt (6 outstanding per wave per step).
static __device__ __forceinline__ void issue_gx_async(_Float16* dstbuf,
                                                      const _Float16* src,
                                                      int tid) {
  unsigned int       l = (unsigned int)(uintptr_t)dstbuf + (unsigned int)tid * 96u;
  unsigned long long gp = (unsigned long long)(uintptr_t)src + (unsigned long long)tid * 96u;
#pragma unroll
  for (int j = 0; j < 6; ++j)
    asm volatile("global_load_async_to_lds_b128 %0, %1, off"
                 :: "v"(l + (unsigned)j * 16u), "v"(gp + (unsigned long long)j * 16u)
                 : "memory");
}

// ---------------------------------------------------------------------------
// Phase 2: persistent per-group recurrent scan, 8 blocks x 256 threads.
// Wave w owns hidden columns [16w,16w+16) -> gate tiles r/z/n element-aligned
// in registers. W_hh fragments live in VGPRs for all 512 steps. gx slabs are
// double-buffered in LDS via async-to-LDS copies pipelined one step ahead.
// ---------------------------------------------------------------------------
__global__ __launch_bounds__(256)
void gru_scan(const float* __restrict__ state,
              const float* __restrict__ W_hh,
              const float* __restrict__ b_hh,
              const _Float16* __restrict__ gx,
              float* __restrict__ y,
              float* __restrict__ h_out) {
  const int g    = blockIdx.x;
  const int tid  = threadIdx.x;
  const int lane = tid & 31;
  const int wave = tid >> 5;
  const int hi   = lane >> 4;
  const int ln   = lane & 15;
  const int c    = wave * 16 + ln;       // owned hidden column

  __shared__ _Float16 h_lds[Bq * HGq];       //  8 KB: hidden state (f16)
  __shared__ _Float16 gxb[2][Bq * NOq];      // 48 KB: double-buffered gx slab

  for (int j = tid; j < Bq * HGq; j += 256)
    h_lds[j] = (_Float16)state[(size_t)g * Bq * HGq + j];

  // Prologue: start async copy of the t=0 slab into buffer 0.
  issue_gx_async(&gxb[0][0], gx + ((size_t)g * Tq) * (Bq * NOq), tid);

  // Register-resident W_hh B-fragments: Bf[gate][kstep]
  v16h Bf[3][4];
  float bia[3];
#pragma unroll
  for (int gate = 0; gate < 3; ++gate) {
    const int o = gate * HGq + c;
    bia[gate] = b_hh[g * NOq + o];
#pragma unroll
    for (int ks = 0; ks < 4; ++ks)
      Bf[gate][ks] = load_bfrag(W_hh + ((size_t)g * NOq + o) * HGq + ks * 32 + hi * 16);
  }
  __syncthreads();

  for (int t = 0; t < Tq; ++t) {
    const int cur = t & 1;

    // Kick off next step's slab into the other buffer (freed by the barrier
    // that ended step t-1: all waves finished reading it).
    if (t + 1 < Tq)
      issue_gx_async(&gxb[cur ^ 1][0],
                     gx + ((size_t)g * Tq + (t + 1)) * (Bq * NOq), tid);

    // A fragments from h (f16 in LDS) + h_old for owned elements.
    v16h Af[2][4];
#pragma unroll
    for (int mt = 0; mt < 2; ++mt)
#pragma unroll
      for (int ks = 0; ks < 4; ++ks) {
        const _Float16* hp = &h_lds[(mt * 16 + ln) * HGq + ks * 32 + hi * 8];
        v16h a;
#pragma unroll
        for (int j = 0; j < 8; ++j) { a[j] = hp[j]; a[j + 8] = hp[16 + j]; }
        Af[mt][ks] = a;
      }
    float hold[2][8];
#pragma unroll
    for (int mt = 0; mt < 2; ++mt)
#pragma unroll
      for (int r = 0; r < 8; ++r)
        hold[mt][r] = (float)h_lds[(mt * 16 + hi * 8 + r) * HGq + c];

    // Drain the async copy that filled gxb[cur] (issued one step ago).
    // Async loads complete in order: <=6 outstanding means the previous
    // step's 6 ops are done.
    if (t + 1 < Tq) asm volatile("s_wait_asynccnt 0x6" ::: "memory");
    else            asm volatile("s_wait_asynccnt 0x0" ::: "memory");
    __syncthreads();   // everyone read h_lds; everyone's slab portion landed

    v8f acc[3][2];
#pragma unroll
    for (int gate = 0; gate < 3; ++gate)
#pragma unroll
      for (int mt = 0; mt < 2; ++mt)
#pragma unroll
        for (int r = 0; r < 8; ++r) acc[gate][mt][r] = bia[gate];

#pragma unroll
    for (int ks = 0; ks < 4; ++ks)
#pragma unroll
      for (int gate = 0; gate < 3; ++gate)
#pragma unroll
        for (int mt = 0; mt < 2; ++mt)
          acc[gate][mt] = __builtin_amdgcn_wmma_f32_16x16x32_f16(
              false, Af[mt][ks], false, Bf[gate][ks], (short)0, acc[gate][mt], false, false);

    // Fused GRU gates, element-aligned across the three gate accumulators.
    const _Float16* gxs = &gxb[cur][0];
#pragma unroll
    for (int mt = 0; mt < 2; ++mt)
#pragma unroll
      for (int r = 0; r < 8; ++r) {
        const int bidx = mt * 16 + hi * 8 + r;
        const int gb = bidx * NOq;
        const float gr = (float)gxs[gb + c];
        const float gz = (float)gxs[gb + HGq + c];
        const float gn = (float)gxs[gb + 2 * HGq + c];
        const float rr = 1.f / (1.f + __expf(-(gr + acc[0][mt][r])));
        const float zz = 1.f / (1.f + __expf(-(gz + acc[1][mt][r])));
        const float ni = gn + rr * acc[2][mt][r];
        const float e2 = __expf(-2.f * ni);
        const float nn = (1.f - e2) / (1.f + e2);     // tanh(ni)
        const float hn = (1.f - zz) * nn + zz * hold[mt][r];
        h_lds[bidx * HGq + c] = (_Float16)hn;
        y[((size_t)bidx * Tq + t) * HIDq + g * HGq + c] = hn;
      }
    __syncthreads();   // h_new + slab reads done before next iteration
  }

  for (int j = tid; j < Bq * HGq; j += 256)
    h_out[(size_t)g * Bq * HGq + j] = (float)h_lds[j];
}

extern "C" void kernel_launch(void* const* d_in, const int* in_sizes, int n_in,
                              void* d_out, int out_size, void* d_ws, size_t ws_size,
                              hipStream_t stream) {
  const float* x   = (const float*)d_in[0];
  const float* st  = (const float*)d_in[1];
  const float* Wih = (const float*)d_in[2];
  const float* Whh = (const float*)d_in[3];
  const float* bih = (const float*)d_in[4];
  const float* bhh = (const float*)d_in[5];

  float* y     = (float*)d_out;
  float* h_out = y + (size_t)Bq * Tq * HIDq;
  _Float16* gx = (_Float16*)d_ws;   // [G][T][B][3*HG] f16 = 100 MB scratch

  hipLaunchKernelGGL(gru_gx_gemm, dim3(64, 8), dim3(256), 0, stream, x, Wih, bih, gx);
  hipLaunchKernelGGL(gru_scan, dim3(8), dim3(256), 0, stream, st, Whh, bhh, gx, y, h_out);
}
